// MultiHeadAttention_52707838656720
// MI455X (gfx1250) — compile-verified, matrix operand order checked by reading, not run on hardware
//
#include <hip/hip_runtime.h>
#include <hip/hip_bf16.h>
#include <stdint.h>

// MI455X / gfx1250: wave32, WMMA bf16 16x16x32, fp32 accumulate.
typedef __attribute__((ext_vector_type(16))) __bf16 v16bf;
typedef __attribute__((ext_vector_type(8)))  __bf16 v8bf;
typedef __attribute__((ext_vector_type(8)))  float  v8f;

#define B_  2
#define S_  2048
#define E_  1024
#define H_  16
#define DH_ 64

// Load one 16-bit A/B WMMA fragment (per-lane 16 bf16 = two 16B chunks 16
// elements apart). Works for both LDS and global pointers.
__device__ __forceinline__ v16bf frag16(const __bf16* p) {
  v8bf lo = *(const v8bf*)(p);
  v8bf hi = *(const v8bf*)(p + 16);
  return __builtin_shufflevector(lo, hi, 0,1,2,3,4,5,6,7,8,9,10,11,12,13,14,15);
}

__device__ __forceinline__ v8f wmma_bf16(v16bf a, v16bf b, v8f c) {
  return __builtin_amdgcn_wmma_f32_16x16x32_bf16(false, a, false, b, (short)0, c,
                                                 false, false);
}

// Online-softmax update over one 32-key block (two 16-key S^T C-tiles).
// Per-lane query = qg; keys for element i are j+khalf+i and j+khalf+i+16.
// MASKED is true only for the single diagonal block of a wave tile.
template <bool MASKED>
__device__ __forceinline__ float softmax_update(v8f& sa, v8f& sb, int j, int khalf,
                                                int qg, float& mrun, float& lrun,
                                                v16bf& pf) {
  const float CEXP = 0.125f * 1.44269504088896f;   // (1/sqrt(Dh)) * log2(e)
  const float NINF = -__builtin_inff();
  float mloc = NINF;
#pragma unroll
  for (int i = 0; i < 8; ++i) {
    if (MASKED) {
      int key0 = j + khalf + i;
      if (key0 > qg)      sa[i] = NINF;
      if (key0 + 16 > qg) sb[i] = NINF;
    }
    mloc = fmaxf(mloc, fmaxf(sa[i], sb[i]));
  }
  mloc = fmaxf(mloc, __shfl_xor(mloc, 16, 32));    // join lane pair (q, q+16)
  float mnew  = fmaxf(mrun, mloc);
  float alpha = __builtin_amdgcn_exp2f((mrun - mnew) * CEXP);
  float lloc = 0.f;
#pragma unroll
  for (int i = 0; i < 8; ++i) {
    float p0 = __builtin_amdgcn_exp2f((sa[i] - mnew) * CEXP);
    float p1 = __builtin_amdgcn_exp2f((sb[i] - mnew) * CEXP);
    lloc += p0 + p1;
    pf[i]     = (__bf16)p0;
    pf[i + 8] = (__bf16)p1;
  }
  lloc += __shfl_xor(lloc, 16, 32);
  lrun = lrun * alpha + lloc;
  mrun = mnew;
  return alpha;
}

// One 32-key attention block for a 32-query wave tile (two 16-query C-tiles
// sharing the K/V fragment stream: 16 WMMA per 16 b128 fragment loads).
template <bool MASKED>
__device__ __forceinline__ void attn_block(
    const __bf16* __restrict__ Kh, const __bf16* __restrict__ Vh, int j,
    int rl16, int khalf, int qg0, int qg1,
    const v16bf& bq00, const v16bf& bq01, const v16bf& bq10, const v16bf& bq11,
    v8f o0[4], v8f o1[4], float& m0, float& l0, float& m1, float& l1) {
  // K fragments for the 32-key block (A operand: M=key, K=d, 2 d-steps).
  const __bf16* kr0 = Kh + (size_t)(j + rl16) * DH_ + khalf;
  const __bf16* kr1 = kr0 + (size_t)16 * DH_;
  v16bf ak00 = frag16(kr0), ak01 = frag16(kr0 + 32);
  v16bf ak10 = frag16(kr1), ak11 = frag16(kr1 + 32);

  // Four 16x16 S^T tiles: {keys j..j+15, j+16..j+31} x {qtile0, qtile1}.
  v8f s00 = {}, s01 = {}, s10 = {}, s11 = {};
  s00 = wmma_bf16(ak00, bq00, s00); s00 = wmma_bf16(ak01, bq01, s00);
  s01 = wmma_bf16(ak10, bq00, s01); s01 = wmma_bf16(ak11, bq01, s01);
  s10 = wmma_bf16(ak00, bq10, s10); s10 = wmma_bf16(ak01, bq11, s10);
  s11 = wmma_bf16(ak10, bq10, s11); s11 = wmma_bf16(ak11, bq11, s11);

  // Online softmax per query sub-tile; P packed as next WMMA B-fragments.
  v16bf pf0, pf1;
  float a0 = softmax_update<MASKED>(s00, s01, j, khalf, qg0, m0, l0, pf0);
  float a1 = softmax_update<MASKED>(s10, s11, j, khalf, qg1, m1, l1, pf1);

#pragma unroll
  for (int nt = 0; nt < 4; ++nt)
#pragma unroll
    for (int i = 0; i < 8; ++i) { o0[nt][i] *= a0; o1[nt][i] *= a1; }

  // O^T += V^T * P^T : one V fragment feeds both query sub-tiles.
#pragma unroll
  for (int nt = 0; nt < 4; ++nt) {
    const __bf16* vr = Vh + (size_t)(16 * nt + rl16) * S_ + j + khalf;
    v16bf av = frag16(vr);
    o0[nt] = wmma_bf16(av, pf0, o0[nt]);
    o1[nt] = wmma_bf16(av, pf1, o1[nt]);
  }
}

// ---------------------------------------------------------------------------
// Kernel 1: fused Q/K/V projection.  out = X @ W^T.
// Block: 256 thr (8 waves), tile 128(M) x 64(N) x 32(K). Waves 4x2, each wave
// 2x2 WMMA tiles. Staging fully unrolled (loads batched before waits).
// Epilogue: Q,K -> [B,H,S,Dh] (scalar scatter); V -> [B,H,Dh,S] (b128 stores,
// since the C-fragment's per-lane M direction is the contiguous s direction).
// ---------------------------------------------------------------------------
__global__ __launch_bounds__(256)
void qkv_proj_kernel(const float* __restrict__ Xq, const float* __restrict__ Xk,
                     const float* __restrict__ Xv,
                     const float* __restrict__ Wq, const float* __restrict__ Wk,
                     const float* __restrict__ Wv,
                     __bf16* __restrict__ Qb, __bf16* __restrict__ Kb,
                     __bf16* __restrict__ Vt) {
  __shared__ __bf16 Al[128 * 40];   // 40 = 32 + 8 pad -> conflict-free b128
  __shared__ __bf16 Bl[64 * 40];

  const int t     = threadIdx.x;
  const int lane  = t & 31;
  const int w     = t >> 5;
  const int wm    = (w & 3) * 32;
  const int wn    = (w >> 2) * 32;
  const int mb    = blockIdx.y * 128;
  const int nb    = blockIdx.x * 64;
  const int mode  = blockIdx.z;                 // 0=Q 1=K 2=V
  const float* X  = (mode == 0) ? Xq : (mode == 1) ? Xk : Xv;
  const float* W  = (mode == 0) ? Wq : (mode == 1) ? Wk : Wv;
  const int rl16  = lane & 15;
  const int khalf = (lane >> 4) << 3;           // 0 or 8

  const int rowA = t >> 3, colA = (t & 7) << 2;        // +32 rows per chunk

  v8f acc[2][2] = {};
  for (int k0 = 0; k0 < E_; k0 += 32) {
    float4 va[4], vb[2];
#pragma unroll
    for (int it = 0; it < 4; ++it)
      va[it] = *(const float4*)(X + (size_t)(mb + rowA + 32 * it) * E_ + k0 + colA);
#pragma unroll
    for (int it = 0; it < 2; ++it)
      vb[it] = *(const float4*)(W + (size_t)(nb + rowA + 32 * it) * E_ + k0 + colA);

    __syncthreads();
#pragma unroll
    for (int it = 0; it < 4; ++it) {
      __bf16* p = &Al[(rowA + 32 * it) * 40 + colA];
      p[0] = (__bf16)va[it].x; p[1] = (__bf16)va[it].y;
      p[2] = (__bf16)va[it].z; p[3] = (__bf16)va[it].w;
    }
#pragma unroll
    for (int it = 0; it < 2; ++it) {
      __bf16* p = &Bl[(rowA + 32 * it) * 40 + colA];
      p[0] = (__bf16)vb[it].x; p[1] = (__bf16)vb[it].y;
      p[2] = (__bf16)vb[it].z; p[3] = (__bf16)vb[it].w;
    }
    __syncthreads();

    v16bf af[2], bfr[2];
#pragma unroll
    for (int mt = 0; mt < 2; ++mt)
      af[mt] = frag16(&Al[(wm + 16 * mt + rl16) * 40 + khalf]);
#pragma unroll
    for (int nt = 0; nt < 2; ++nt)
      bfr[nt] = frag16(&Bl[(wn + 16 * nt + rl16) * 40 + khalf]);
#pragma unroll
    for (int mt = 0; mt < 2; ++mt)
#pragma unroll
      for (int nt = 0; nt < 2; ++nt)
        acc[mt][nt] = wmma_bf16(af[mt], bfr[nt], acc[mt][nt]);
  }

  // Epilogue (once per 1024-deep K reduction).
  if (mode == 2) {
    // V: C per-lane M direction == contiguous s of Vt[b][h][d][s].
#pragma unroll
    for (int mt = 0; mt < 2; ++mt)
#pragma unroll
      for (int nt = 0; nt < 2; ++nt) {
        int m0v = mb + wm + 16 * mt + khalf;        // 8 consecutive tokens
        int n   = nb + wn + 16 * nt + rl16;         // feature
        int b = m0v >> 11, s0 = m0v & (S_ - 1);
        int h = n >> 6,    d  = n & (DH_ - 1);
        v8bf ov;
#pragma unroll
        for (int i = 0; i < 8; ++i) ov[i] = (__bf16)acc[mt][nt][i];
        *(v8bf*)(Vt + (((size_t)b * H_ + h) * DH_ + d) * S_ + s0) = ov;
      }
  } else {
    __bf16* Oqk = (mode == 0) ? Qb : Kb;
#pragma unroll
    for (int mt = 0; mt < 2; ++mt)
#pragma unroll
      for (int nt = 0; nt < 2; ++nt)
#pragma unroll
        for (int i = 0; i < 8; ++i) {
          int m = mb + wm + 16 * mt + khalf + i;    // C: M = i + 8*(lane>=16)
          int n = nb + wn + 16 * nt + rl16;         // C: N = lane & 15
          int b = m >> 11, s = m & (S_ - 1);
          int h = n >> 6,  d = n & (DH_ - 1);
          Oqk[(((size_t)b * H_ + h) * S_ + s) * DH_ + d] = (__bf16)acc[mt][nt][i];
        }
  }
}

// ---------------------------------------------------------------------------
// Kernel 2: causal flash attention, one wave per 32-query tile.
// S^T = K*Q^T trick: score C-fragments are already the B-fragment layout for
// O^T = V^T*P^T (zero cross-lane transpose); softmax stats per-lane (+1
// shfl_xor16). Only the diagonal block needs masking -> mask-free main loop.
// ---------------------------------------------------------------------------
__global__ __launch_bounds__(256)
void attn_kernel(const __bf16* __restrict__ Qb, const __bf16* __restrict__ Kb,
                 const __bf16* __restrict__ Vt, __bf16* __restrict__ Ob) {
  const int t    = threadIdx.x;
  const int lane = t & 31;
  const int wave = t >> 5;
  const int tile = blockIdx.x * 8 + wave;        // B*H*(S/32) = 2048 tiles
  const int b    = tile >> 10;
  const int h    = (tile >> 6) & (H_ - 1);
  const int q0   = (tile & 63) << 5;             // 32 queries per wave
  const int rl16  = lane & 15;
  const int khalf = (lane >> 4) << 3;
  const int qg0   = q0 + rl16;                   // query row, C-tile 0
  const int qg1   = q0 + 16 + rl16;              // query row, C-tile 1
  const float NINF = -__builtin_inff();

  const __bf16* Qh = Qb + ((size_t)b * H_ + h) * S_ * DH_;
  const __bf16* Kh = Kb + ((size_t)b * H_ + h) * S_ * DH_;
  const __bf16* Vh = Vt + ((size_t)b * H_ + h) * DH_ * S_;

  // Resident Q (B-operand) fragments for both query sub-tiles.
  const __bf16* qrow0 = Qh + (size_t)(q0 + rl16) * DH_ + khalf;
  const __bf16* qrow1 = qrow0 + (size_t)16 * DH_;
  const v16bf bq00 = frag16(qrow0), bq01 = frag16(qrow0 + 32);
  const v16bf bq10 = frag16(qrow1), bq11 = frag16(qrow1 + 32);

  v8f o0[4] = {}, o1[4] = {};                     // O^T accumulators (Dh=64)
  float m0 = NINF, l0 = 0.f, m1 = NINF, l1 = 0.f;

  // Interior blocks: max key = j+31 <= q0 <= min query  ->  no masking.
  for (int j = 0; j < q0; j += 32)
    attn_block<false>(Kh, Vh, j, rl16, khalf, qg0, qg1,
                      bq00, bq01, bq10, bq11, o0, o1, m0, l0, m1, l1);
  // Single diagonal block (j == q0): causal masking applies.
  attn_block<true>(Kh, Vh, q0, rl16, khalf, qg0, qg1,
                   bq00, bq01, bq10, bq11, o0, o1, m0, l0, m1, l1);

  // Normalize and store O as bf16 [B,S,E]: 8 consecutive d per lane per tile.
  float r0 = 1.0f / l0, r1 = 1.0f / l1;
  __bf16* orow0 = Ob + ((size_t)b * S_ + qg0) * E_ + h * DH_ + khalf;
  __bf16* orow1 = Ob + ((size_t)b * S_ + qg1) * E_ + h * DH_ + khalf;
#pragma unroll
  for (int nt = 0; nt < 4; ++nt) {
    v8bf ov0, ov1;
#pragma unroll
    for (int i = 0; i < 8; ++i) {
      ov0[i] = (__bf16)(o0[nt][i] * r0);
      ov1[i] = (__bf16)(o1[nt][i] * r1);
    }
    *(v8bf*)(orow0 + 16 * nt) = ov0;
    *(v8bf*)(orow1 + 16 * nt) = ov1;
  }
}

// ---------------------------------------------------------------------------
// Kernel 3: output projection  out = O @ Wo^T + bo  (fp32 result).
// Flipped orientation: A = Wo rows (M = feature), B = O rows (N = token), so
// each lane's 8 C elements are 8 consecutive output floats -> float4 stores.
// ---------------------------------------------------------------------------
__global__ __launch_bounds__(256)
void out_proj_kernel(const __bf16* __restrict__ Ob, const float* __restrict__ Wo,
                     const float* __restrict__ bo, float* __restrict__ out) {
  __shared__ __bf16 Al[128 * 40];   // Wo tile: 128 features x 32 k
  __shared__ __bf16 Bl[64 * 40];    // O  tile: 64 tokens   x 32 k

  const int t     = threadIdx.x;
  const int lane  = t & 31;
  const int w     = t >> 5;
  const int wm    = (w & 3) * 32;
  const int wn    = (w >> 2) * 32;
  const int mb    = blockIdx.y * 128;             // feature base
  const int nb    = blockIdx.x * 64;              // token base
  const int rl16  = lane & 15;
  const int khalf = (lane >> 4) << 3;

  const int rowA = t >> 3, colA = (t & 7) << 2;   // fp32: 4-elem chunks
  const int rowB = t >> 2, colB = (t & 3) << 3;   // bf16: 8-elem chunks

  v8f acc[2][2] = {};
  for (int k0 = 0; k0 < E_; k0 += 32) {
    float4 va[4];
    v8bf vb;
#pragma unroll
    for (int it = 0; it < 4; ++it)
      va[it] = *(const float4*)(Wo + (size_t)(mb + rowA + 32 * it) * E_ + k0 + colA);
    vb = *(const v8bf*)(Ob + (size_t)(nb + rowB) * E_ + k0 + colB);

    __syncthreads();
#pragma unroll
    for (int it = 0; it < 4; ++it) {
      __bf16* p = &Al[(rowA + 32 * it) * 40 + colA];
      p[0] = (__bf16)va[it].x; p[1] = (__bf16)va[it].y;
      p[2] = (__bf16)va[it].z; p[3] = (__bf16)va[it].w;
    }
    *(v8bf*)(&Bl[rowB * 40 + colB]) = vb;
    __syncthreads();

    v16bf af[2], bfr[2];
#pragma unroll
    for (int mt = 0; mt < 2; ++mt)
      af[mt] = frag16(&Al[(wm + 16 * mt + rl16) * 40 + khalf]);
#pragma unroll
    for (int nt = 0; nt < 2; ++nt)
      bfr[nt] = frag16(&Bl[(wn + 16 * nt + rl16) * 40 + khalf]);
#pragma unroll
    for (int mt = 0; mt < 2; ++mt)
#pragma unroll
      for (int nt = 0; nt < 2; ++nt)
        acc[mt][nt] = wmma_bf16(af[mt], bfr[nt], acc[mt][nt]);
  }

  // Epilogue: C[M=feature, N=token] -> out[token][feature] vectorized.
#pragma unroll
  for (int mt = 0; mt < 2; ++mt)
#pragma unroll
    for (int nt = 0; nt < 2; ++nt) {
      int nf0  = mb + wm + 16 * mt + khalf;       // 8 consecutive features
      int mtok = nb + wn + 16 * nt + rl16;        // token
      float* po = out + (size_t)mtok * E_ + nf0;
      const float4 b0 = *(const float4*)(bo + nf0);
      const float4 b1 = *(const float4*)(bo + nf0 + 4);
      float4 lo, hi;
      lo.x = acc[mt][nt][0] + b0.x; lo.y = acc[mt][nt][1] + b0.y;
      lo.z = acc[mt][nt][2] + b0.z; lo.w = acc[mt][nt][3] + b0.w;
      hi.x = acc[mt][nt][4] + b1.x; hi.y = acc[mt][nt][5] + b1.y;
      hi.z = acc[mt][nt][6] + b1.z; hi.w = acc[mt][nt][7] + b1.w;
      *(float4*)(po)     = lo;
      *(float4*)(po + 4) = hi;
    }
}

// ---------------------------------------------------------------------------
extern "C" void kernel_launch(void* const* d_in, const int* in_sizes, int n_in,
                              void* d_out, int out_size, void* d_ws, size_t ws_size,
                              hipStream_t stream) {
  (void)in_sizes; (void)n_in; (void)out_size; (void)ws_size;
  const float* q  = (const float*)d_in[0];
  const float* k  = (const float*)d_in[1];
  const float* v  = (const float*)d_in[2];
  // d_in[3] is the causal mask -- causality is baked into the kernel.
  const float* Wq = (const float*)d_in[4];
  const float* Wk = (const float*)d_in[5];
  const float* Wv = (const float*)d_in[6];
  const float* Wo = (const float*)d_in[7];
  const float* bo = (const float*)d_in[8];

  const size_t nQKV = (size_t)B_ * H_ * S_ * DH_;     // 4M bf16 elems each
  __bf16* Qb = (__bf16*)d_ws;
  __bf16* Kb = Qb + nQKV;
  __bf16* Vt = Kb + nQKV;
  __bf16* Ob = Vt + nQKV;                             // total ws: 32 MB

  dim3 gp(E_ / 64, (B_ * S_) / 128, 3);
  qkv_proj_kernel<<<gp, dim3(256), 0, stream>>>(q, k, v, Wq, Wk, Wv, Qb, Kb, Vt);

  dim3 ga((B_ * H_ * (S_ / 32)) / 8);
  attn_kernel<<<ga, dim3(256), 0, stream>>>(Qb, Kb, Vt, Ob);

  dim3 go((B_ * S_) / 64, E_ / 128);
  out_proj_kernel<<<go, dim3(256), 0, stream>>>(Ob, Wo, bo, (float*)d_out);
}
